// mLSTMCell_52312701665693
// MI455X (gfx1250) — matrix operand order repaired and checked by reading, not compiled, not run on hardware
//
#include <hip/hip_runtime.h>
#include <hip/hip_bf16.h>

// ---------------------------------------------------------------------------
// mLSTM cell for MI455X (gfx1250, wave32)
//   Phase 0: f32 -> bf16 conversion of x and all weight matrices
//   Phase 1: WMMA GEMMs (v_wmma_f32_16x16x32_bf16):
//            - o / kvq projections: 64x64 per-wave register blocking
//              (4x4 accumulators -> 1 b128 fragment load per WMMA)
//            - tiny i/f projection (N=16): one wave per 16x16 tile
//            Fused epilogues: sigmoid(o), k/sqrt(HS)+b, v+b, q+b, exp(i/f)
//   Phase 2: sequential (c,n) scan; one 256-thread block per (b,h) chain,
//            c state (128x128 f32) register-resident (64 f32 per thread)
// ---------------------------------------------------------------------------

typedef __bf16 bf16_t;
typedef __attribute__((ext_vector_type(8)))  __bf16 v8bf;
typedef __attribute__((ext_vector_type(16))) __bf16 v16bf;
typedef __attribute__((ext_vector_type(8)))  float  v8f;

static constexpr int kB   = 8;
static constexpr int kT   = 2048;
static constexpr int kD   = 1024;
static constexpr int kH   = 8;
static constexpr int kHS  = 128;
static constexpr int kBT  = kB * kT;        // 16384 rows
static constexpr int kNo  = kH * kHS;       // 1024
static constexpr int kNkq = kH * 3 * kHS;   // 3072
static constexpr int kNif = kH * 2;         // 16

// ---------------------------------------------------------------- convert ---
__global__ void f32_to_bf16_kernel(const float* __restrict__ src,
                                   bf16_t* __restrict__ dst, int n) {
  int i = blockIdx.x * blockDim.x + threadIdx.x;
  int stride = gridDim.x * blockDim.x;
  for (; i < n; i += stride) dst[i] = (bf16_t)src[i];
}

// --------------------------------------------------------- WMMA fragments ---
// A (16x32 bf16): lane l holds row m=l&15; K = e + (l>=16?16:0)
//   -> one contiguous 32B run at p (p already includes lhalf*16).
__device__ __forceinline__ v16bf load_frag_a(const bf16_t* p) {
  v8bf lo = *(const v8bf*)(p);
  v8bf hi = *(const v8bf*)(p + 8);
  return __builtin_shufflevector(lo, hi,
      0,1,2,3,4,5,6,7,8,9,10,11,12,13,14,15);
}
// B (32x16 bf16): lane l holds col n=l&15; K = e + (l>=16?16:0)
//   -> one contiguous 32B run at p (p already includes lhalf*16).
__device__ __forceinline__ v16bf load_frag_b(const bf16_t* p) {
  v8bf lo = *(const v8bf*)(p);
  v8bf hi = *(const v8bf*)(p + 8);
  return __builtin_shufflevector(lo, hi,
      0,1,2,3,4,5,6,7,8,9,10,11,12,13,14,15);
}

__device__ __forceinline__ float epilogue(float v, int n, int mode,
                                          const float* __restrict__ bias) {
  if (mode == 0) {                      // o gate: sigmoid(acc + bias)
    v += bias[n];
    return 1.0f / (1.0f + __expf(-v));
  } else if (mode == 1) {               // kvq
    const int h = n / 384, e = n % 384;
    const int sub = e >> 7, ei = e & 127;
    if (sub == 0) return v * 0.08838834764831845f + bias[h * 384 + ei];
    return v + bias[h * 384 + (sub << 7) + ei];
  }
  return __expf(v + bias[n]);           // i/f gates: pre-exponentiate
}

// ------------------------------------------------- blocked GEMM (N >= 64) ---
// Y(M=kBT, N) = Xbf(M,K) * Wbf(N,K)^T. One wave per 64x64 tile:
// 4x4 grid of 16x16x32 WMMA accumulators, 16 WMMAs per 16 b128 loads.
__global__ __launch_bounds__(256)
void gemm_wmma_blk_kernel(const bf16_t* __restrict__ A,   // (kBT, kD)
                          const bf16_t* __restrict__ W,   // (N,  kD)
                          const float*  __restrict__ bias,
                          float* __restrict__ Y, int N, int mode) {
  const int wid   = blockIdx.x * (blockDim.x >> 5) + (threadIdx.x >> 5);
  const int lane  = threadIdx.x & 31;
  const int ntile = N >> 6;
  const int m0 = (wid / ntile) << 6;
  const int n0 = (wid % ntile) << 6;
  const int lhalf = lane >> 4;
  const int lmod  = lane & 15;

  const bf16_t* abase = A + (size_t)(m0 + lmod) * kD + lhalf * 16;
  const bf16_t* bbase = W + (size_t)(n0 + lmod) * kD + lhalf * 16;

  v8f acc[4][4];
  #pragma unroll
  for (int ai = 0; ai < 4; ++ai)
    #pragma unroll
    for (int bi = 0; bi < 4; ++bi) acc[ai][bi] = (v8f){};

  for (int k = 0; k < kD; k += 32) {
    v16bf afrag[4], bfrag[4];
    #pragma unroll
    for (int i = 0; i < 4; ++i)
      afrag[i] = load_frag_a(abase + (size_t)(i * 16) * kD + k);
    #pragma unroll
    for (int i = 0; i < 4; ++i)
      bfrag[i] = load_frag_b(bbase + (size_t)(i * 16) * kD + k);
    #pragma unroll
    for (int ai = 0; ai < 4; ++ai)
      #pragma unroll
      for (int bi = 0; bi < 4; ++bi)
        acc[ai][bi] = __builtin_amdgcn_wmma_f32_16x16x32_bf16(
            false, afrag[ai], false, bfrag[bi],
            (short)0, acc[ai][bi], false, false);
  }

  #pragma unroll
  for (int ai = 0; ai < 4; ++ai) {
    #pragma unroll
    for (int bi = 0; bi < 4; ++bi) {
      const int n = n0 + (bi << 4) + lmod;
      #pragma unroll
      for (int j = 0; j < 8; ++j) {
        const int m = m0 + (ai << 4) + j + (lhalf << 3);
        Y[(size_t)m * N + n] = epilogue(acc[ai][bi][j], n, mode, bias);
      }
    }
  }
}

// -------------------------------------------------- small GEMM (N == 16) ----
__global__ __launch_bounds__(256)
void gemm_wmma_kernel(const bf16_t* __restrict__ A,
                      const bf16_t* __restrict__ W,
                      const float*  __restrict__ bias,
                      float* __restrict__ Y, int N, int mode) {
  const int wid   = blockIdx.x * (blockDim.x >> 5) + (threadIdx.x >> 5);
  const int lane  = threadIdx.x & 31;
  const int ntile = N >> 4;
  const int m0 = (wid / ntile) << 4;
  const int n0 = (wid % ntile) << 4;
  const int lhalf = lane >> 4;
  const int lmod  = lane & 15;

  const bf16_t* arow = A + (size_t)(m0 + lmod) * kD + lhalf * 16;
  const bf16_t* brow = W + (size_t)(n0 + lmod) * kD + lhalf * 16;

  v8f acc = {};
  for (int k = 0; k < kD; k += 32) {
    v16bf af = load_frag_a(arow + k);
    v16bf bfrag = load_frag_b(brow + k);
    acc = __builtin_amdgcn_wmma_f32_16x16x32_bf16(
        false, af, false, bfrag, (short)0, acc, false, false);
  }

  const int n = n0 + lmod;
  #pragma unroll
  for (int j = 0; j < 8; ++j) {
    const int m = m0 + j + (lhalf << 3);
    Y[(size_t)m * N + n] = epilogue(acc[j], n, mode, bias);
  }
}

// ------------------------------------------------------------------- scan ---
// One block per (b,h). 256 threads: tid -> (half = tid>>7, row r = tid&127).
// Thread owns c[r][half*64 .. half*64+63] in registers.
__global__ __launch_bounds__(256)
void mlstm_scan_kernel(const float* __restrict__ ifg,    // (kBT, 16)  exp'ed
                       const float* __restrict__ ogate,  // (kBT, 1024) sigmoid'ed
                       const float* __restrict__ kvq,    // (kBT, 3072) biased
                       float* __restrict__ out) {        // (kBT, 1024)
  __shared__ __align__(16) float sK[kHS];
  __shared__ __align__(16) float sQ[kHS];
  __shared__ __align__(16) float sV[kHS];
  __shared__ __align__(16) float sO[kHS];
  __shared__ float sPart[256];
  __shared__ float sWP[4];

  const int bh = blockIdx.x;          // 0..63
  const int b = bh >> 3, h = bh & 7;
  const int tid = threadIdx.x;
  const int half = tid >> 7;
  const int r = tid & 127;
  const int colbase = half << 6;      // 0 or 64
  const int wv = tid >> 5, lane = tid & 31;

  float creg[64];
  #pragma unroll
  for (int j = 0; j < 64; ++j) creg[j] = 0.f;
  float nreg = 0.f;

  for (int t = 0; t < kT; ++t) {
    const size_t base = (size_t)(b * kT + t);
    const size_t kvb = base * kNkq + h * 384;
    if (half == 0) {
      sK[r] = kvq[kvb + r];
      sQ[r] = kvq[kvb + 256 + r];
    } else {
      sV[r] = kvq[kvb + 128 + r];
      sO[r] = ogate[base * kNo + (h << 7) + r];
    }
    const float it = ifg[base * kNif + (h << 1) + 0];
    const float ft = ifg[base * kNif + (h << 1) + 1];
    __syncthreads();

    // n-state update + denom partial (rows live in half-0 threads)
    if (half == 0) {
      nreg = ft * nreg + it * sK[r];
      float prod = nreg * sQ[r];
      #pragma unroll
      for (int off = 16; off > 0; off >>= 1) prod += __shfl_down(prod, off, 32);
      if (lane == 0) sWP[wv] = prod;
    }

    // c-state update + matvec partial: broadcast b128 LDS reads of k,q
    const float a = it * sV[r];
    float hp = 0.f;
    const float4* k4 = (const float4*)(sK + colbase);
    const float4* q4 = (const float4*)(sQ + colbase);
    #pragma unroll
    for (int j4 = 0; j4 < 16; ++j4) {
      const float4 kk = k4[j4];
      const float4 qq = q4[j4];
      creg[4*j4+0] = ft * creg[4*j4+0] + a * kk.x; hp = fmaf(creg[4*j4+0], qq.x, hp);
      creg[4*j4+1] = ft * creg[4*j4+1] + a * kk.y; hp = fmaf(creg[4*j4+1], qq.y, hp);
      creg[4*j4+2] = ft * creg[4*j4+2] + a * kk.z; hp = fmaf(creg[4*j4+2], qq.z, hp);
      creg[4*j4+3] = ft * creg[4*j4+3] + a * kk.w; hp = fmaf(creg[4*j4+3], qq.w, hp);
    }
    sPart[tid] = hp;
    __syncthreads();

    if (half == 0) {
      float denom = sWP[0] + sWP[1] + sWP[2] + sWP[3];
      denom = fmaxf(fabsf(denom), 1.0f);
      const float hv = sPart[r] + sPart[r + 128];
      out[base * kNo + (h << 7) + r] = sO[r] * hv / denom;
    }
    __syncthreads();   // protect sK..sO from next iteration's staging
  }
}

// ----------------------------------------------------------------- launch ---
extern "C" void kernel_launch(void* const* d_in, const int* in_sizes, int n_in,
                              void* d_out, int out_size, void* d_ws, size_t ws_size,
                              hipStream_t stream) {
  const float* x      = (const float*)d_in[0];   // (8,2048,1024)
  const float* if_w   = (const float*)d_in[1];   // (8,2,1024)   -> (16,1024)
  const float* o_w    = (const float*)d_in[2];   // (8,128,1024) -> (1024,1024)
  const float* kvq_w  = (const float*)d_in[3];   // (8,384,1024) -> (3072,1024)
  const float* if_b   = (const float*)d_in[4];   // (16)
  const float* o_b    = (const float*)d_in[5];   // (1024)
  const float* kvq_b  = (const float*)d_in[6];   // (3072)
  float* out = (float*)d_out;

  // workspace carve-out (256B aligned)
  char* ws = (char*)d_ws;
  size_t off = 0;
  auto carve = [&](size_t bytes) -> void* {
    void* p = ws + off;
    off += (bytes + 255) & ~(size_t)255;
    return p;
  };
  bf16_t* xbf    = (bf16_t*)carve((size_t)kBT * kD * sizeof(bf16_t));
  bf16_t* owbf   = (bf16_t*)carve((size_t)kNo * kD * sizeof(bf16_t));
  bf16_t* kvqwbf = (bf16_t*)carve((size_t)kNkq * kD * sizeof(bf16_t));
  bf16_t* ifwbf  = (bf16_t*)carve((size_t)kNif * kD * sizeof(bf16_t));
  float*  ogate  = (float*)carve((size_t)kBT * kNo * sizeof(float));
  float*  kvqact = (float*)carve((size_t)kBT * kNkq * sizeof(float));
  float*  ifgact = (float*)carve((size_t)kBT * kNif * sizeof(float));

  // Phase 0: f32 -> bf16
  f32_to_bf16_kernel<<<4096, 256, 0, stream>>>(x,     xbf,    kBT * kD);
  f32_to_bf16_kernel<<<1024, 256, 0, stream>>>(o_w,   owbf,   kNo * kD);
  f32_to_bf16_kernel<<<2048, 256, 0, stream>>>(kvq_w, kvqwbf, kNkq * kD);
  f32_to_bf16_kernel<<<  64, 256, 0, stream>>>(if_w,  ifwbf,  kNif * kD);

  // Phase 1: WMMA projections
  {
    // 64x64 per-wave tiles, 8 waves per block
    const int tiles_o   = (kBT / 64) * (kNo / 64);    // 256*16  = 4096
    const int tiles_kvq = (kBT / 64) * (kNkq / 64);   // 256*48  = 12288
    gemm_wmma_blk_kernel<<<tiles_o / 8,   256, 0, stream>>>(xbf, owbf,   o_b,   ogate,  kNo,  0);
    gemm_wmma_blk_kernel<<<tiles_kvq / 8, 256, 0, stream>>>(xbf, kvqwbf, kvq_b, kvqact, kNkq, 1);
    // tiny N=16 GEMM: one wave per 16x16 tile
    const int tiles_if = (kBT / 16) * (kNif / 16);    // 1024
    gemm_wmma_kernel<<<tiles_if / 8, 256, 0, stream>>>(xbf, ifwbf, if_b, ifgact, kNif, 2);
  }

  // Phase 2: sequential scan, one block per (b,h)
  mlstm_scan_kernel<<<kB * kH, 256, 0, stream>>>(ifgact, ogate, kvqact, out);
}
